// EulerianGNN_55173149884915
// MI455X (gfx1250) — matrix operand order repaired and checked by reading, not compiled, not run on hardware
//
#include <hip/hip_runtime.h>
#include <hip/hip_bf16.h>

// ---------------------------------------------------------------------------
// EulerianGNN for MI455X (gfx1250): bf16 WMMA GEMMs with LDS-staged weights,
// software-pipelined DS fragment loads, f32 LN/softmax/residual epilogues.
// ---------------------------------------------------------------------------

typedef __bf16 bf16_t;
typedef __attribute__((ext_vector_type(16))) __bf16 bf16x16;
typedef __attribute__((ext_vector_type(8)))  __bf16 bf16x8;
typedef __attribute__((ext_vector_type(8)))  float  f32x8;

#define GEMM_WAVES 8   // waves (m-tiles) per workgroup

static __device__ __forceinline__ float clamp11(float x) {
    return fminf(fmaxf(x, -1.0f), 1.0f);
}

static __device__ __forceinline__ void atomicMaxF(float* addr, float val) {
    unsigned int* a = (unsigned int*)addr;
    unsigned int old = __float_as_uint(*addr);
    while (__uint_as_float(old) < val) {
        unsigned int assumed = old;
        old = atomicCAS(a, assumed, __float_as_uint(val));
        if (old == assumed) break;
    }
}

// ---------------------------------------------------------------------------
// Generic WMMA GEMM: C[M,128] = act( A[M,K](bf16, optionally row-gathered)
//                                    @ Bcm[128,K](bf16 col-major) + bias/Cin )
// 8 waves per block; B panel staged once into LDS (padded stride K+8 so b128
// DS reads are uniform 2-request, conflict-free); each wave owns a 16-row
// tile and produces all 128 output columns (8 f32 accumulator fragments).
// B fragments are double-buffered in VGPRs so DS latency overlaps WMMA issue.
// ---------------------------------------------------------------------------
__global__ __launch_bounds__(256)
void wmma_gemm128(const bf16_t* __restrict__ A, int lda, const int* __restrict__ rowmap,
                  const bf16_t* __restrict__ B, int ldb, int kb0,
                  const float* __restrict__ bias, const float* __restrict__ Cin,
                  float* __restrict__ outF, bf16_t* __restrict__ outB,
                  int M, int K, int relu)
{
    extern __shared__ char smem_raw[];
    bf16_t* Bs = (bf16_t*)smem_raw;          // [128][K+8]
    const int K8 = K + 8;                    // padded LDS row stride (bf16 elems)

    // ---- cooperative stage of the B panel into LDS (16B chunks)
    {
        const int chunksPerCol = K >> 3;         // K multiple of 32
        const int total = 128 * chunksPerCol;
        for (int i = threadIdx.x; i < total; i += blockDim.x) {
            int col = i / chunksPerCol, ch = i - col * chunksPerCol;
            *(bf16x8*)(Bs + (size_t)col * K8 + ch * 8) =
                *(const bf16x8*)(B + (size_t)col * ldb + kb0 + ch * 8);
        }
    }
    __syncthreads();

    const int mtiles = (M + 15) >> 4;
    const int wave   = blockIdx.x * GEMM_WAVES + (threadIdx.x >> 5);
    if (wave >= mtiles) return;

    const int lane = threadIdx.x & 31;
    const int n    = lane & 15;   // column-within-16 / A-row-within-16
    const int hi   = lane >> 4;   // half-wave

    // A row this lane loads (A fragment rows are indexed by lane&15)
    int row_a = wave * 16 + n;
    int ra    = row_a < M ? row_a : (M - 1);
    if (rowmap) ra = rowmap[ra];
    const bf16_t* Arow = A + (size_t)ra * lda;

    f32x8 acc[8];
#pragma unroll
    for (int t = 0; t < 8; ++t) {
#pragma unroll
        for (int j = 0; j < 8; ++j) {
            int r = wave * 16 + hi * 8 + j;
            int c = t * 16 + n;
            float v = 0.0f;
            if (Cin)      { if (r < M) v = Cin[(size_t)r * 128 + c]; }
            else if (bias)  v = bias[c];
            acc[t][j] = v;
        }
    }

    // B 32x16 fragment from LDS: lane = column (t*16+n); half -> K 0..15 / 16..31
    auto loadB = [&](int t, int kt) -> bf16x16 {
        const bf16_t* Bcol = Bs + (size_t)(t * 16 + n) * K8 + kt * 32 + hi * 16;
        bf16x8 b0 = *(const bf16x8*)Bcol;
        bf16x8 b1 = *(const bf16x8*)(Bcol + 8);
        bf16x16 r;
#pragma unroll
        for (int i = 0; i < 8; ++i) { r[i] = b0[i]; r[8 + i] = b1[i]; }
        return r;
    };

    const int nk = K >> 5;
    for (int kt = 0; kt < nk; ++kt) {
        // 16-bit A 16x32 fragment: half 0 -> K 0..15; half 1 -> K 16..31
        bf16x16 afrag;
        {
            bf16x8 lo = *(const bf16x8*)(Arow + kt * 32 + hi * 16);
            bf16x8 up = *(const bf16x8*)(Arow + kt * 32 + hi * 16 + 8);
#pragma unroll
            for (int i = 0; i < 8; ++i) { afrag[i] = lo[i]; afrag[8 + i] = up[i]; }
        }
        // double-buffered B fragments: DS load for t+1 in flight during WMMA(t)
        bf16x16 bcur = loadB(0, kt);
#pragma unroll
        for (int t = 0; t < 7; ++t) {
            bf16x16 bnxt = loadB(t + 1, kt);
            acc[t] = __builtin_amdgcn_wmma_f32_16x16x32_bf16(
                false, afrag, false, bcur, (short)0, acc[t], false, false);
            bcur = bnxt;
        }
        acc[7] = __builtin_amdgcn_wmma_f32_16x16x32_bf16(
            false, afrag, false, bcur, (short)0, acc[7], false, false);
    }

#pragma unroll
    for (int t = 0; t < 8; ++t) {
#pragma unroll
        for (int j = 0; j < 8; ++j) {
            int r = wave * 16 + hi * 8 + j;
            if (r >= M) continue;
            int c = t * 16 + n;
            float v = acc[t][j];
            if (relu) v = v > 0.0f ? v : 0.0f;
            if (outF) outF[(size_t)r * 128 + c] = v;
            if (outB) outB[(size_t)r * 128 + c] = (bf16_t)v;
        }
    }
}

// ---------------------------------------------------------------------------
// Elementwise / helper kernels
// ---------------------------------------------------------------------------
__global__ void pack_weight(const float* __restrict__ W, bf16_t* __restrict__ out,
                            int K, int Kpad)
{
    int id = blockIdx.x * blockDim.x + threadIdx.x;
    if (id >= 128 * Kpad) return;
    int col = id / Kpad, k = id % Kpad;
    out[(size_t)col * Kpad + k] = (k < K) ? (bf16_t)W[(size_t)k * 128 + col] : (bf16_t)0.0f;
}

__global__ void fill_f32(float* __restrict__ p, long long n, float v)
{
    long long id = (long long)blockIdx.x * blockDim.x + threadIdx.x;
    if (id < n) p[id] = v;
}

__global__ void cvt_bf16(const float* __restrict__ in, bf16_t* __restrict__ out, long long n)
{
    long long id = (long long)blockIdx.x * blockDim.x + threadIdx.x;
    if (id < n) out[id] = (bf16_t)in[id];
}

__global__ void node_feats(const float* __restrict__ field, const float* __restrict__ pos,
                           const int* __restrict__ bct, const float* __restrict__ bc_emb,
                           bf16_t* __restrict__ out, int N)
{
    int nn = blockIdx.x * blockDim.x + threadIdx.x;
    if (nn >= N) return;
    bf16_t* o = out + (size_t)nn * 32;
    const float inv = 1.0f / (1.0f + 1e-8f);
    for (int i = 0; i < 5; ++i) o[i] = (bf16_t)(field[(size_t)nn * 5 + i] * inv);
    float px = pos[(size_t)nn * 2 + 0], py = pos[(size_t)nn * 2 + 1];
    o[5] = (bf16_t)clamp11(px / 10.0f);           // (pos - lo)/room
    o[6] = (bf16_t)clamp11(py / 2.0f);
    o[7] = (bf16_t)clamp11((10.0f - px) / 10.0f); // (hi - pos)/room
    o[8] = (bf16_t)clamp11((2.0f - py) / 2.0f);
    int b = bct[nn];
    for (int j = 0; j < 8; ++j) o[9 + j] = (bf16_t)bc_emb[b * 8 + j];
    for (int j = 17; j < 32; ++j) o[j] = (bf16_t)0.0f;
}

__global__ void edge_feats(const float* __restrict__ pos, const int* __restrict__ ei,
                           const float* __restrict__ fno, const float* __restrict__ far_,
                           const int* __restrict__ fty, const float* __restrict__ fte,
                           bf16_t* __restrict__ out, int E)
{
    int e = blockIdx.x * blockDim.x + threadIdx.x;
    if (e >= E) return;
    int s = ei[e], d = ei[E + e];
    float dx = pos[(size_t)d * 2 + 0] - pos[(size_t)s * 2 + 0];
    float dy = pos[(size_t)d * 2 + 1] - pos[(size_t)s * 2 + 1];
    float dist = sqrtf(dx * dx + dy * dy);
    dist = fmaxf(dist, 1e-8f);
    bf16_t* o = out + (size_t)e * 32;
    o[0] = (bf16_t)fno[(size_t)e * 2 + 0];
    o[1] = (bf16_t)fno[(size_t)e * 2 + 1];
    o[2] = (bf16_t)far_[e];       // / L_REF^2 with L_REF = 1
    o[3] = (bf16_t)dist;          // / L_REF
    o[4] = (bf16_t)(dx / dist);
    o[5] = (bf16_t)(dy / dist);
    int t = fty[e];
    for (int j = 0; j < 8; ++j) o[6 + j] = (bf16_t)fte[t * 8 + j];
    for (int j = 14; j < 32; ++j) o[j] = (bf16_t)0.0f;
}

// LayerNorm over 128, then optional residual update + bf16 refreshes.
__global__ void ln_update(const float* __restrict__ pre, const float* __restrict__ g,
                          const float* __restrict__ beta,
                          float* __restrict__ resid, float* __restrict__ outF,
                          bf16_t* __restrict__ obf_ln, bf16_t* __restrict__ obf_res, int M)
{
    int r = blockIdx.x * blockDim.x + threadIdx.x;
    if (r >= M) return;
    const float* x = pre + (size_t)r * 128;
    float m = 0.0f;
    for (int c = 0; c < 128; ++c) m += x[c];
    m *= (1.0f / 128.0f);
    float v = 0.0f;
    for (int c = 0; c < 128; ++c) { float d = x[c] - m; v += d * d; }
    v *= (1.0f / 128.0f);
    float inv = rsqrtf(v + 1e-5f);
    for (int c = 0; c < 128; ++c) {
        float ln = (x[c] - m) * inv * g[c] + beta[c];
        size_t idx = (size_t)r * 128 + c;
        if (outF)   outF[idx] = ln;
        if (obf_ln) obf_ln[idx] = (bf16_t)ln;
        if (resid) {
            float t = resid[idx] + ln;
            resid[idx] = t;
            if (obf_res) obf_res[idx] = (bf16_t)t;
        } else if (obf_res) {
            obf_res[idx] = (bf16_t)ln;
        }
    }
}

__global__ void attn_scores(const float* __restrict__ Qn, const float* __restrict__ Kn,
                            const int* __restrict__ ei, float* __restrict__ sc,
                            float* __restrict__ smax, int E)
{
    long long id = (long long)blockIdx.x * blockDim.x + threadIdx.x;
    if (id >= (long long)E * 4) return;
    int e = (int)(id >> 2), h = (int)(id & 3);
    int s = ei[e], d = ei[E + e];
    const float* q = Qn + (size_t)d * 128 + h * 32;
    const float* k = Kn + (size_t)s * 128 + h * 32;
    float acc = 0.0f;
    for (int i = 0; i < 32; ++i) acc += q[i] * k[i];
    acc *= 0.17677669529663687f;  // 1/sqrt(32)
    sc[id] = acc;
    atomicMaxF(&smax[(size_t)d * 4 + h], acc);
}

__global__ void attn_expsum(float* __restrict__ sc, const float* __restrict__ smax,
                            float* __restrict__ ssum, const int* __restrict__ ei, int E)
{
    long long id = (long long)blockIdx.x * blockDim.x + threadIdx.x;
    if (id >= (long long)E * 4) return;
    int e = (int)(id >> 2), h = (int)(id & 3);
    int d = ei[E + e];
    float es = expf(sc[id] - smax[(size_t)d * 4 + h]);
    sc[id] = es;
    atomicAdd(&ssum[(size_t)d * 4 + h], es);
}

__global__ void attn_scatter(const float* __restrict__ sc, const float* __restrict__ ssum,
                             const float* __restrict__ V, const int* __restrict__ ei,
                             float* __restrict__ msg, int E)
{
    long long id = (long long)blockIdx.x * blockDim.x + threadIdx.x;
    if (id >= (long long)E * 128) return;
    int e = (int)(id >> 7), c = (int)(id & 127);
    int h = c >> 5;
    int d = ei[E + e];
    float alpha = sc[(size_t)e * 4 + h] / (ssum[(size_t)d * 4 + h] + 1e-12f);
    atomicAdd(&msg[(size_t)d * 128 + c], alpha * V[id]);
}

__global__ void dec_final(const bf16_t* __restrict__ hbf, const float* __restrict__ W,
                          const float* __restrict__ b, float* __restrict__ out, int N)
{
    long long id = (long long)blockIdx.x * blockDim.x + threadIdx.x;
    if (id >= (long long)N * 5) return;
    int nn = (int)(id / 5), o = (int)(id % 5);
    float s = b[o];
    const bf16_t* h = hbf + (size_t)nn * 128;
    for (int k = 0; k < 128; ++k) s += (float)h[k] * W[(size_t)k * 5 + o];
    out[id] = s;
}

// ---------------------------------------------------------------------------
// Host side
// ---------------------------------------------------------------------------
extern "C" void kernel_launch(void* const* d_in, const int* in_sizes, int n_in,
                              void* d_out, int out_size, void* d_ws, size_t ws_size,
                              hipStream_t stream)
{
    (void)n_in; (void)out_size; (void)ws_size;

    const int N = in_sizes[0] / 5;   // field [N,5]
    const int E = in_sizes[5];       // face_areas [E]

    // ---- input pointers (top-level: dict insertion order; params: pytree sorted keys)
    const float* field  = (const float*)d_in[0];
    const float* pos    = (const float*)d_in[1];
    const int*   bct    = (const int*)  d_in[2];
    const int*   ei     = (const int*)  d_in[3];   // [2,E]: src then dst
    const float* fno    = (const float*)d_in[4];
    const float* far_   = (const float*)d_in[5];
    const int*   fty    = (const int*)  d_in[6];
    const float* bc_emb = (const float*)d_in[7];
    // blocks: base 8 (block0) / 28 (block1): W_K,W_O,W_Q,W_V, edge_mlp(8), node_mlp(8)
    // decoder 48..53, edge_encoder 54..61, face_type_embed 62, node_encoder 63..70
    const float* fte    = (const float*)d_in[62];

    auto P = [&](int i) { return (const float*)d_in[i]; };

    // ---- workspace bump allocator
    char* wp = (char*)d_ws;
    auto alloc = [&](size_t bytes) -> void* {
        void* p = (void*)wp;
        wp += (bytes + 255) & ~(size_t)255;
        return p;
    };
    auto aB = [&](size_t elems) { return (bf16_t*)alloc(elems * sizeof(bf16_t)); };
    auto aF = [&](size_t elems) { return (float*) alloc(elems * sizeof(float)); };

    // packed weights (bf16 col-major [128][Kpad])
    bf16_t* pw_neW0 = aB(128 * 32);  bf16_t* pw_neW1 = aB(128 * 128); bf16_t* pw_neW2 = aB(128 * 128);
    bf16_t* pw_eeW0 = aB(128 * 32);  bf16_t* pw_eeW1 = aB(128 * 128); bf16_t* pw_eeW2 = aB(128 * 128);
    bf16_t* pw_WK[2], *pw_WO[2], *pw_WQ[2], *pw_WV[2];
    bf16_t* pw_eW0[2], *pw_eW1[2], *pw_eW2[2];
    bf16_t* pw_nW0[2], *pw_nW1[2], *pw_nW2[2];
    for (int b = 0; b < 2; ++b) {
        pw_WK[b] = aB(128 * 128); pw_WO[b] = aB(128 * 128);
        pw_WQ[b] = aB(128 * 128); pw_WV[b] = aB(128 * 128);
        pw_eW0[b] = aB(128 * 384); pw_eW1[b] = aB(128 * 128); pw_eW2[b] = aB(128 * 128);
        pw_nW0[b] = aB(128 * 256); pw_nW1[b] = aB(128 * 128); pw_nW2[b] = aB(128 * 128);
    }
    bf16_t* pw_dW0 = aB(128 * 128); bf16_t* pw_dW1 = aB(128 * 128);

    // activations
    bf16_t* ndfeat  = aB((size_t)N * 32);
    bf16_t* edfeat  = aB((size_t)E * 32);
    bf16_t* x_bf    = aB((size_t)N * 128);
    bf16_t* e_bf    = aB((size_t)E * 128);
    bf16_t* enew_bf = aB((size_t)E * 128);
    bf16_t* hE1     = aB((size_t)E * 128);
    bf16_t* hE2     = aB((size_t)E * 128);
    bf16_t* hN1     = aB((size_t)N * 128);
    bf16_t* hN2     = aB((size_t)N * 128);
    float*  x_f32   = aF((size_t)N * 128);
    float*  e_f32   = aF((size_t)E * 128);
    float*  ebuf    = aF((size_t)E * 128);   // edge-size f32 scratch (pre-LN / V)
    float*  nbuf    = aF((size_t)N * 128);   // node-size f32 scratch
    float*  Qn      = aF((size_t)N * 128);
    float*  Kn      = aF((size_t)N * 128);
    float*  smax    = aF((size_t)N * 4);
    float*  ssum    = aF((size_t)N * 4);
    float*  exps    = aF((size_t)E * 4);
    float*  msg     = aF((size_t)N * 128);

    // ---- launch helpers
    auto gemm = [&](const bf16_t* A, int lda, const int* rowmap,
                    const bf16_t* B, int ldb, int kb0,
                    const float* bias, const float* Cin,
                    float* outF, bf16_t* outB, int M, int K, int relu) {
        int mtiles = (M + 15) / 16;
        int blocks = (mtiles + GEMM_WAVES - 1) / GEMM_WAVES;
        size_t shmem = (size_t)128 * (K + 8) * sizeof(bf16_t);
        wmma_gemm128<<<blocks, GEMM_WAVES * 32, shmem, stream>>>(
            A, lda, rowmap, B, ldb, kb0, bias, Cin, outF, outB, M, K, relu);
    };
    auto pack = [&](const float* W, bf16_t* out, int K, int Kpad) {
        int total = 128 * Kpad;
        pack_weight<<<(total + 255) / 256, 256, 0, stream>>>(W, out, K, Kpad);
    };
    auto fill = [&](float* p, long long n, float v) {
        fill_f32<<<(unsigned)((n + 255) / 256), 256, 0, stream>>>(p, n, v);
    };

    // ---- pack all weights
    pack(P(63), pw_neW0, 17, 32); pack(P(65), pw_neW1, 128, 128); pack(P(67), pw_neW2, 128, 128);
    pack(P(54), pw_eeW0, 14, 32); pack(P(56), pw_eeW1, 128, 128); pack(P(58), pw_eeW2, 128, 128);
    for (int b = 0; b < 2; ++b) {
        int base = 8 + b * 20;
        pack(P(base + 0), pw_WK[b], 128, 128);
        pack(P(base + 1), pw_WO[b], 128, 128);
        pack(P(base + 2), pw_WQ[b], 128, 128);
        pack(P(base + 3), pw_WV[b], 128, 128);
        pack(P(base + 4),  pw_eW0[b], 384, 384);
        pack(P(base + 6),  pw_eW1[b], 128, 128);
        pack(P(base + 8),  pw_eW2[b], 128, 128);
        pack(P(base + 12), pw_nW0[b], 256, 256);
        pack(P(base + 14), pw_nW1[b], 128, 128);
        pack(P(base + 16), pw_nW2[b], 128, 128);
    }
    pack(P(48), pw_dW0, 128, 128); pack(P(50), pw_dW1, 128, 128);

    // ---- features
    node_feats<<<(N + 255) / 256, 256, 0, stream>>>(field, pos, bct, bc_emb, ndfeat, N);
    edge_feats<<<(E + 255) / 256, 256, 0, stream>>>(pos, ei, fno, far_, fty, fte, edfeat, E);

    // ---- node encoder: 32(pad17)->128 relu ->128 relu ->128 -> LN
    gemm(ndfeat, 32, nullptr, pw_neW0, 32, 0, P(64), nullptr, nullptr, hN1, N, 32, 1);
    gemm(hN1, 128, nullptr, pw_neW1, 128, 0, P(66), nullptr, nullptr, hN2, N, 128, 1);
    gemm(hN2, 128, nullptr, pw_neW2, 128, 0, P(68), nullptr, nbuf, nullptr, N, 128, 0);
    ln_update<<<(N + 255) / 256, 256, 0, stream>>>(nbuf, P(69), P(70),
                                                   nullptr, x_f32, x_bf, nullptr, N);

    // ---- edge encoder
    gemm(edfeat, 32, nullptr, pw_eeW0, 32, 0, P(55), nullptr, nullptr, hE1, E, 32, 1);
    gemm(hE1, 128, nullptr, pw_eeW1, 128, 0, P(57), nullptr, nullptr, hE2, E, 128, 1);
    gemm(hE2, 128, nullptr, pw_eeW2, 128, 0, P(59), nullptr, ebuf, nullptr, E, 128, 0);
    ln_update<<<(E + 255) / 256, 256, 0, stream>>>(ebuf, P(60), P(61),
                                                   nullptr, e_f32, e_bf, nullptr, E);

    const int* src = ei;
    const int* dst = ei + E;

    // ---- transformer blocks
    for (int b = 0; b < 2; ++b) {
        int base = 8 + b * 20;
        const float *eb0 = P(base + 5), *eb1 = P(base + 7), *eb2 = P(base + 9);
        const float *eg = P(base + 10), *ebeta = P(base + 11);
        const float *nb0 = P(base + 13), *nb1 = P(base + 15), *nb2 = P(base + 17);
        const float *ng = P(base + 18), *nbeta = P(base + 19);

        fill(smax, (long long)N * 4, -3.0e38f);
        fill(ssum, (long long)N * 4, 0.0f);
        fill(msg,  (long long)N * 128, 0.0f);

        // edge_mlp layer1 = gathered concat GEMM: [x[dst] | x[src] | e] @ W0(384,128)
        gemm(x_bf, 128, dst, pw_eW0[b], 384, 0,   eb0, nullptr, ebuf, nullptr, E, 128, 0);
        gemm(x_bf, 128, src, pw_eW0[b], 384, 128, nullptr, ebuf, ebuf, nullptr, E, 128, 0);
        gemm(e_bf, 128, nullptr, pw_eW0[b], 384, 256, nullptr, ebuf, nullptr, hE1, E, 128, 1);
        gemm(hE1, 128, nullptr, pw_eW1[b], 128, 0, eb1, nullptr, nullptr, hE2, E, 128, 1);
        gemm(hE2, 128, nullptr, pw_eW2[b], 128, 0, eb2, nullptr, ebuf, nullptr, E, 128, 0);
        // e_new = LN(ebuf); e += e_new (f32 + bf16 refresh); enew_bf = bf16(e_new)
        ln_update<<<(E + 255) / 256, 256, 0, stream>>>(ebuf, eg, ebeta,
                                                       e_f32, nullptr, enew_bf, e_bf, E);

        // per-node Q,K projections; per-edge V = e_new @ W_V (f32 into ebuf)
        gemm(x_bf, 128, nullptr, pw_WQ[b], 128, 0, nullptr, nullptr, Qn, nullptr, N, 128, 0);
        gemm(x_bf, 128, nullptr, pw_WK[b], 128, 0, nullptr, nullptr, Kn, nullptr, N, 128, 0);
        gemm(enew_bf, 128, nullptr, pw_WV[b], 128, 0, nullptr, nullptr, ebuf, nullptr, E, 128, 0);

        long long e4 = (long long)E * 4, e128 = (long long)E * 128;
        attn_scores <<<(unsigned)((e4 + 255) / 256), 256, 0, stream>>>(Qn, Kn, ei, exps, smax, E);
        attn_expsum <<<(unsigned)((e4 + 255) / 256), 256, 0, stream>>>(exps, smax, ssum, ei, E);
        attn_scatter<<<(unsigned)((e128 + 255) / 256), 256, 0, stream>>>(exps, ssum, ebuf, ei, msg, E);

        // msgO = (segment_sum) @ W_O  (bf16 out)
        cvt_bf16<<<(unsigned)(((long long)N * 128 + 255) / 256), 256, 0, stream>>>(msg, hN2, (long long)N * 128);
        gemm(hN2, 128, nullptr, pw_WO[b], 128, 0, nullptr, nullptr, nullptr, hN1, N, 128, 0);

        // node_mlp: [x | msgO] @ W0(256,128) relu -> 128 relu -> 128 -> LN; x += LN
        gemm(x_bf, 128, nullptr, pw_nW0[b], 256, 0,   nb0, nullptr, nbuf, nullptr, N, 128, 0);
        gemm(hN1, 128, nullptr, pw_nW0[b], 256, 128, nullptr, nbuf, nullptr, hN2, N, 128, 1);
        gemm(hN2, 128, nullptr, pw_nW1[b], 128, 0, nb1, nullptr, nullptr, hN1, N, 128, 1);
        gemm(hN1, 128, nullptr, pw_nW2[b], 128, 0, nb2, nullptr, nbuf, nullptr, N, 128, 0);
        ln_update<<<(N + 255) / 256, 256, 0, stream>>>(nbuf, ng, nbeta,
                                                       x_f32, nullptr, nullptr, x_bf, N);
    }

    // ---- decoder: 128 relu -> 128 relu -> 5 (+bias, no LN)
    gemm(x_bf, 128, nullptr, pw_dW0, 128, 0, P(49), nullptr, nullptr, hN1, N, 128, 1);
    gemm(hN1, 128, nullptr, pw_dW1, 128, 0, P(51), nullptr, nullptr, hN2, N, 128, 1);
    dec_final<<<(unsigned)(((long long)N * 5 + 255) / 256), 256, 0, stream>>>(
        hN2, P(52), P(53), (float*)d_out, N);
}
